// Attention_TopK_70076686402466
// MI455X (gfx1250) — compile-verified, matrix operand order checked by reading, not run on hardware
//
#include <hip/hip_runtime.h>
#include <hip/hip_bf16.h>
#include <stdint.h>

// ---------------------------------------------------------------------------
// Problem constants (reference: B=64, N=197, C=768, H=12, hd=64)
// ---------------------------------------------------------------------------
#define BATCH   64
#define SEQ     197
#define CH      768
#define HEADS   12
#define HDIM    64
#define NPAD    224            // pad seq to 7*32 for the P@V K-loop
#define ROWS    (BATCH * SEQ)  // 12608 = 788 * 16 exactly
#define RT_M    (ROWS / 16)    // 788 row tiles
#define KEEP    137            // int(0.7 * 196)
#define CLS_N   196
#define CLS_PAD 256

typedef __attribute__((ext_vector_type(16))) _Float16 v16h;
typedef __attribute__((ext_vector_type(8)))  _Float16 h8;
typedef __attribute__((ext_vector_type(4)))  _Float16 h4;
typedef __attribute__((ext_vector_type(8)))  float    v8f;
typedef __attribute__((ext_vector_type(4)))  float    f4;

// ---------------------------------------------------------------------------
// WMMA helper: D = A(16x32 f16) * B(32x16 f16) + C(16x16 f32)
// ---------------------------------------------------------------------------
__device__ __forceinline__ v8f wmma_f16(v16h a, v16h b, v8f c) {
  return __builtin_amdgcn_wmma_f32_16x16x32_f16(
      /*neg_a=*/false, a, /*neg_b=*/false, b,
      /*c_mod=*/(short)0, c, /*reuse_a=*/false, /*reuse_b=*/false);
}

// Fragment loader for f16 memory with per-lane contiguous K.
// Per ISA 7.12.2: lane holds K = kb..kb+7 in elems 0..7 and K = kb+16..kb+23
// in elems 8..15, where kb = (lane>>4)*8; caller passes p = row_base + kb.
// Works for A fragments (row-major M x K) and for B fragments when the B
// source is laid out [N][K] (i.e. B^T row-major).
__device__ __forceinline__ v16h ldA16(const _Float16* p) {
  h8 lo = *(const h8*)(p);
  h8 hi = *(const h8*)(p + 16);
  v16h r;
#pragma unroll
  for (int i = 0; i < 8; ++i) { r[i] = lo[i]; r[8 + i] = hi[i]; }
  return r;
}

// ---------------------------------------------------------------------------
// Kernel 0a: zero scratch (qkv pad rows/cols + cls accumulator); grid-stride.
// ---------------------------------------------------------------------------
__global__ __launch_bounds__(256) void zero_kernel(uint32_t* p, size_t n32) {
  size_t i = (size_t)blockIdx.x * blockDim.x + threadIdx.x;
  size_t stride = (size_t)gridDim.x * blockDim.x;
  for (; i < n32; i += stride) p[i] = 0u;
}

// ---------------------------------------------------------------------------
// Kernel 0b: one-shot f32 -> f16 conversion (x, qkv_w, proj_w), 4-wide.
// ---------------------------------------------------------------------------
__global__ __launch_bounds__(256) void cvt_kernel(const float* __restrict__ src,
                                                  _Float16* __restrict__ dst,
                                                  size_t n4) {
  size_t i = (size_t)blockIdx.x * blockDim.x + threadIdx.x;
  size_t stride = (size_t)gridDim.x * blockDim.x;
  for (; i < n4; i += stride) {
    f4 v = ((const f4*)src)[i];
    h4 h;
#pragma unroll
    for (int j = 0; j < 4; ++j) h[j] = (_Float16)v[j];
    ((h4*)dst)[i] = h;
  }
}

// ---------------------------------------------------------------------------
// Kernel 1: QKV GEMM. qkv = x @ qkv_w.T, all-f16 operands.
// One wave computes a 16x64 tile (4 independent accumulators sharing one A
// fragment per K-chunk -> 4 back-to-back independent WMMAs, 96 per wave).
// Each 64-col block is exactly one head of one of q/k/v.
// q,k stored [B,H,NPAD,HDIM]; v stored TRANSPOSED [B,H,HDIM,NPAD] so that
// the attention P@V B-fragments are contiguous loads.
// ---------------------------------------------------------------------------
__global__ __launch_bounds__(32) void qkv_kernel(const _Float16* __restrict__ x16,
                                                 const _Float16* __restrict__ w16,
                                                 _Float16* __restrict__ qws,
                                                 _Float16* __restrict__ kws,
                                                 _Float16* __restrict__ vws_t) {
  const int rt = blockIdx.x;            // 0..787
  const int ct = blockIdx.y;            // 0..35 (64-col blocks over 2304)
  const int lane = threadIdx.x;
  const int m = lane & 15;
  const int half = lane >> 4;
  const int kb = half * 8;

  const _Float16* arow = x16 + (size_t)(rt * 16 + m) * CH;
  const _Float16* brow[4];
#pragma unroll
  for (int s = 0; s < 4; ++s)
    brow[s] = w16 + (size_t)(ct * 64 + s * 16 + m) * CH;  // qkv_w[col][k]

  v8f acc[4] = {v8f{}, v8f{}, v8f{}, v8f{}};
#pragma unroll 2
  for (int kc = 0; kc < CH; kc += 32) {
    __builtin_prefetch(arow + kc + 512, 0, 3);
    v16h a = ldA16(arow + kc + kb);
#pragma unroll
    for (int s = 0; s < 4; ++s) {
      v16h b = ldA16(brow[s] + kc + kb);
      acc[s] = wmma_f16(a, b, acc[s]);
    }
  }

  // Epilogue: which matrix + head are constant per block.
  const int which = ct / HEADS;         // 0=q 1=k 2=v
  const int h     = ct % HEADS;
#pragma unroll
  for (int s = 0; s < 4; ++s) {
    const int d = s * 16 + m;
#pragma unroll
    for (int r = 0; r < 8; ++r) {
      const int grow = rt * 16 + r + 8 * half;  // < 12608 always
      const int b = grow / SEQ;
      const int n = grow % SEQ;
      const _Float16 val = (_Float16)acc[s][r];
      if (which == 2)
        vws_t[((size_t)(b * HEADS + h) * HDIM + d) * NPAD + n] = val;
      else if (which == 1)
        kws[((size_t)(b * HEADS + h) * NPAD + n) * HDIM + d] = val;
      else
        qws[((size_t)(b * HEADS + h) * NPAD + n) * HDIM + d] = val;
    }
  }
}

// ---------------------------------------------------------------------------
// Kernel 2: fused attention. One wave per (b, h, row-tile of 16).
// S = (q@k^T)*scale in LDS -> softmax -> P f16 -> O = P@Vt -> attn-out f16.
// Row 0 of rt==0 accumulates cls attention (head-sum; top-k scale-invariant).
// ---------------------------------------------------------------------------
__global__ __launch_bounds__(32) void attn_kernel(const _Float16* __restrict__ q,
                                                  const _Float16* __restrict__ k,
                                                  const _Float16* __restrict__ vt,
                                                  _Float16* __restrict__ o,
                                                  float* __restrict__ cls) {
  __shared__ float    S[16][NPAD];        // 14 KB
  __shared__ _Float16 P[16][NPAD + 8];    // 7.25 KB, row stride 232 (16B mult)

  const int bid = blockIdx.x;
  const int rt = bid % 13;
  const int h  = (bid / 13) % HEADS;
  const int b  = bid / (13 * HEADS);
  const int lane = threadIdx.x;
  const int m = lane & 15;
  const int half = lane >> 4;
  const int kb = half * 8;

  const _Float16* qb = q  + (size_t)(b * HEADS + h) * NPAD * HDIM;
  const _Float16* kp = k  + (size_t)(b * HEADS + h) * NPAD * HDIM;
  const _Float16* vb = vt + (size_t)(b * HEADS + h) * HDIM * NPAD;

  // Q fragments for this 16-row tile (K = 64 -> two 32-chunks).
  const int qrow = rt * 16 + m;           // < 208 <= NPAD
  v16h a0 = ldA16(qb + (size_t)qrow * HDIM + 0  + kb);
  v16h a1 = ldA16(qb + (size_t)qrow * HDIM + 32 + kb);

  // ---- S = q @ k^T, scaled; 14 col tiles cover NPAD ----
  const float scale = 0.125f;             // hd^-0.5
#pragma unroll 2
  for (int ct = 0; ct < NPAD / 16; ++ct) {
    const int cb = ct * 16;
    // k layout [n][hd]: B(K=kk, N=n) = k[cb+n][kk] -> contiguous in kk.
    v16h b0 = ldA16(kp + (size_t)(cb + m) * HDIM + 0  + kb);
    v16h b1 = ldA16(kp + (size_t)(cb + m) * HDIM + 32 + kb);
    v8f acc = {};
    acc = wmma_f16(a0, b0, acc);
    acc = wmma_f16(a1, b1, acc);
#pragma unroll
    for (int r = 0; r < 8; ++r)
      S[r + 8 * half][cb + m] = acc[r] * scale;
  }
  __syncthreads();

  // ---- softmax per row (lanes 0..15 each own one row) ----
  if (lane < 16) {
    float mx = -1e30f;
    for (int c = 0; c < SEQ; ++c) mx = fmaxf(mx, S[lane][c]);
    float sum = 0.f;
    for (int c = 0; c < SEQ; ++c) {
      float e = __expf(S[lane][c] - mx);
      S[lane][c] = e;
      sum += e;
    }
    const float inv = 1.0f / sum;
    for (int c = 0; c < NPAD; ++c) {
      float pv = (c < SEQ) ? S[lane][c] * inv : 0.f;
      if (c < SEQ) S[lane][c] = pv;
      P[lane][c] = (_Float16)pv;
    }
  }
  __syncthreads();

  // cls-token attention accumulation (mean over heads = sum / const).
  if (rt == 0) {
    for (int c = lane; c < CLS_N; c += 32)
      atomicAdd(&cls[b * CLS_PAD + c], S[0][c + 1]);
  }
  __syncthreads();

  // ---- O = P @ V : K = NPAD = 7 chunks of 32; 4 output col tiles of 16.
  // Vt layout [d][n]: B(K=kk, N=d) = Vt[d][kk] -> contiguous in kk.
  v8f oacc[4] = {v8f{}, v8f{}, v8f{}, v8f{}};
#pragma unroll 2
  for (int kc = 0; kc < NPAD / 32; ++kc) {
    const int kbase = kc * 32;
    v16h ap = ldA16(&P[m][kbase + kb]);
#pragma unroll
    for (int dt = 0; dt < 4; ++dt) {
      v16h bv = ldA16(vb + (size_t)(dt * 16 + m) * NPAD + kbase + kb);
      oacc[dt] = wmma_f16(ap, bv, oacc[dt]);
    }
  }

  // Store attn-out f16 as [B*SEQ][CH] rows for the proj GEMM.
#pragma unroll
  for (int dt = 0; dt < 4; ++dt) {
#pragma unroll
    for (int r = 0; r < 8; ++r) {
      const int orow = rt * 16 + r + 8 * half;
      if (orow < SEQ)
        o[((size_t)(b * SEQ + orow)) * CH + h * HDIM + dt * 16 + m] =
            (_Float16)oacc[dt][r];
    }
  }
}

// ---------------------------------------------------------------------------
// Kernel 3: Proj GEMM + bias. out = O @ proj_w.T + b (f32 to d_out).
// 16x64 tile per wave, all-f16 operands.
// ---------------------------------------------------------------------------
__global__ __launch_bounds__(32) void proj_kernel(const _Float16* __restrict__ o,
                                                  const _Float16* __restrict__ w16,
                                                  const float* __restrict__ bias,
                                                  float* __restrict__ out) {
  const int rt = blockIdx.x;            // 0..787
  const int ct = blockIdx.y;            // 0..11 (64-col blocks over 768)
  const int lane = threadIdx.x;
  const int m = lane & 15;
  const int half = lane >> 4;
  const int kb = half * 8;

  const _Float16* arow = o + (size_t)(rt * 16 + m) * CH;
  const _Float16* brow[4];
#pragma unroll
  for (int s = 0; s < 4; ++s)
    brow[s] = w16 + (size_t)(ct * 64 + s * 16 + m) * CH;

  v8f acc[4] = {v8f{}, v8f{}, v8f{}, v8f{}};
#pragma unroll 2
  for (int kc = 0; kc < CH; kc += 32) {
    __builtin_prefetch(arow + kc + 512, 0, 3);
    v16h a = ldA16(arow + kc + kb);
#pragma unroll
    for (int s = 0; s < 4; ++s) {
      v16h b = ldA16(brow[s] + kc + kb);
      acc[s] = wmma_f16(a, b, acc[s]);
    }
  }

#pragma unroll
  for (int s = 0; s < 4; ++s) {
    const int col = ct * 64 + s * 16 + m;
    const float bv = bias[col];
#pragma unroll
    for (int r = 0; r < 8; ++r) {
      const int grow = rt * 16 + r + 8 * half;
      out[(size_t)grow * CH + col] = acc[s][r] + bv;
    }
  }
}

// ---------------------------------------------------------------------------
// Kernel 4: top-137 of 196 per batch (sorted desc, ties -> lower index),
// then write idx (float) and broadcast index [137, 768] (float).
// ---------------------------------------------------------------------------
__global__ __launch_bounds__(256) void topk_kernel(const float* __restrict__ cls,
                                                   float* __restrict__ index_out,
                                                   float* __restrict__ idx_out) {
  __shared__ float sval[CLS_PAD];
  __shared__ float rv[256];
  __shared__ int   ri[256];
  __shared__ int   ids[KEEP];

  const int b = blockIdx.x;
  const int t = threadIdx.x;

  sval[t] = (t < CLS_N) ? cls[b * CLS_PAD + t] : -1e30f;
  __syncthreads();

  for (int it = 0; it < KEEP; ++it) {
    rv[t] = sval[t];
    ri[t] = t;
    __syncthreads();
#pragma unroll
    for (int s = 128; s > 0; s >>= 1) {
      if (t < s) {
        float v2 = rv[t + s]; int i2 = ri[t + s];
        if (v2 > rv[t] || (v2 == rv[t] && i2 < ri[t])) { rv[t] = v2; ri[t] = i2; }
      }
      __syncthreads();
    }
    if (t == 0) { ids[it] = ri[0]; sval[ri[0]] = -1e30f; }
    __syncthreads();
  }

  for (int j = t; j < KEEP; j += 256)
    idx_out[b * KEEP + j] = (float)ids[j];
  for (int e = t; e < KEEP * CH; e += 256)
    index_out[(size_t)b * KEEP * CH + e] = (float)ids[e / CH];
}

// ---------------------------------------------------------------------------
// Launch
// ---------------------------------------------------------------------------
extern "C" void kernel_launch(void* const* d_in, const int* in_sizes, int n_in,
                              void* d_out, int out_size, void* d_ws, size_t ws_size,
                              hipStream_t stream) {
  const float* x      = (const float*)d_in[0];
  const float* qkv_w  = (const float*)d_in[1];
  const float* proj_w = (const float*)d_in[2];
  const float* proj_b = (const float*)d_in[3];

  // Workspace layout (bytes): q | k | vT | cls | o | x16 | w16 | pw16
  const size_t QSZ = (size_t)BATCH * HEADS * NPAD * HDIM * sizeof(_Float16); // 22,020,096
  const size_t CLS_BYTES = (size_t)BATCH * CLS_PAD * sizeof(float);          // 65,536
  const size_t OSZ  = (size_t)ROWS * CH * sizeof(_Float16);                  // 19,365,888
  const size_t X16SZ = OSZ;
  const size_t W16SZ = (size_t)3 * CH * CH * sizeof(_Float16);               // 3,538,944
  char* ws = (char*)d_ws;
  _Float16* q_ws   = (_Float16*)(ws);
  _Float16* k_ws   = (_Float16*)(ws + QSZ);
  _Float16* vt_ws  = (_Float16*)(ws + 2 * QSZ);
  float*    cls_ws = (float*)   (ws + 3 * QSZ);
  _Float16* o_ws   = (_Float16*)(ws + 3 * QSZ + CLS_BYTES);
  _Float16* x16_ws = (_Float16*)(ws + 3 * QSZ + CLS_BYTES + OSZ);
  _Float16* w16_ws = (_Float16*)(ws + 3 * QSZ + CLS_BYTES + OSZ + X16SZ);
  _Float16* pw16_ws= (_Float16*)(ws + 3 * QSZ + CLS_BYTES + OSZ + X16SZ + W16SZ);

  // d_out layout (floats): out [B,N,C] | index [B,137,C] | idx [B,137]
  float* out_ptr   = (float*)d_out;
  float* index_ptr = out_ptr + (size_t)BATCH * SEQ * CH;
  float* idx_ptr   = index_ptr + (size_t)BATCH * KEEP * CH;

  // 0a) zero q/k/vT (pad rows/cols must be 0) + cls accumulator
  const size_t n32 = (3 * QSZ + CLS_BYTES) / 4;
  zero_kernel<<<4096, 256, 0, stream>>>((uint32_t*)d_ws, n32);

  // 0b) one-shot f16 conversions
  cvt_kernel<<<2048, 256, 0, stream>>>(x,      x16_ws,  (size_t)ROWS * CH / 4);
  cvt_kernel<<<1024, 256, 0, stream>>>(qkv_w,  w16_ws,  (size_t)3 * CH * CH / 4);
  cvt_kernel<<<512,  256, 0, stream>>>(proj_w, pw16_ws, (size_t)CH * CH / 4);

  // 1) qkv projection (16x64 tiles)
  qkv_kernel<<<dim3(RT_M, 36), 32, 0, stream>>>(x16_ws, w16_ws, q_ws, k_ws, vt_ws);

  // 2) fused attention
  attn_kernel<<<BATCH * HEADS * 13, 32, 0, stream>>>(q_ws, k_ws, vt_ws, o_ws, cls_ws);

  // 3) output projection + bias (16x64 tiles)
  proj_kernel<<<dim3(RT_M, 12), 32, 0, stream>>>(o_ws, pw16_ws, proj_b, out_ptr);

  // 4) top-k token selection
  topk_kernel<<<BATCH, 256, 0, stream>>>(cls_ws, index_ptr, idx_ptr);
}
